// Qwen3Attention_27668179321538
// MI455X (gfx1250) — matrix operand order repaired and checked by reading, not compiled, not run on hardware
//
#include <hip/hip_runtime.h>
#include <hip/hip_bf16.h>
#include <math.h>

// ---------------------------------------------------------------------------
// Qwen3 GQA attention block for MI455X (gfx1250, wave32, WMMA bf16 path,
// async VMEM->LDS staging for V tiles)
// ---------------------------------------------------------------------------

typedef __attribute__((ext_vector_type(16))) __bf16          v16bf;
typedef __attribute__((ext_vector_type(8)))  float           v8f;
typedef __attribute__((ext_vector_type(16))) unsigned short  v16u;

#define B_      2
#define L_      2048
#define D_      2048
#define H_      16
#define KVH_    4
#define HD_     128
#define GROUPS_ (H_ / KVH_)
#define SCALE_  0.08838834764831845f   // 128^-0.5
#define NEG_    (-1000000000.0f)
#define EPS_    1e-6f

#define VSTRIDE_ 136   // V tile LDS row stride (elements); 136*2B = 272B = 17*16B

// fp32 -> bf16 (round to nearest even)
static __device__ __forceinline__ unsigned short f2bf(float f) {
  unsigned u = __builtin_bit_cast(unsigned, f);
  unsigned r = u + 0x7FFFu + ((u >> 16) & 1u);
  return (unsigned short)(r >> 16);
}

// A-fragment K index for 16-bit 16x32 WMMA A layout (ISA 7.12.2):
// VGPR j holds K pair {k,k+1}; lanes 16-31 have K offset +8.
static __device__ __forceinline__ int kfA(int j, int half) {
  return (j < 4 ? j * 2 : 16 + (j - 4) * 2) + half * 8;
}

static __device__ __forceinline__ v8f zero8() {
  v8f z;
#pragma unroll
  for (int i = 0; i < 8; i++) z[i] = 0.0f;
  return z;
}

// ---------------------------------------------------------------------------
// Kernel 1/4 (x4 uses): C[M,N] = A[M,K] (fp32) * W[N,K]^T (fp32), bf16 WMMA.
// 128x128 block tile, K-step 32, 8 waves (2x4), each wave 64x32 -> 8 wmma/step.
// (Staging stays on the VALU path: it performs the fp32->bf16 conversion.)
// ---------------------------------------------------------------------------
__global__ __launch_bounds__(256) void gemm_xwT(const float* __restrict__ A,
                                                const float* __restrict__ W,
                                                float* __restrict__ C,
                                                int M, int N, int K) {
  __shared__ unsigned short As[128 * 36];   // [m][k] k-stride 36 (pad: no bank conflicts)
  __shared__ unsigned short Wt[32 * 132];   // [k][n] n-stride 132 (transposed for B-frag pairs)

  const int tid   = threadIdx.x;
  const int lane  = tid & 31;
  const int wvid  = tid >> 5;
  const int half  = lane >> 4;
  const int ln    = lane & 15;
  const int waveM = wvid & 1;   // 2 waves along M
  const int waveN = wvid >> 1;  // 4 waves along N
  const int m0    = blockIdx.y * 128;
  const int n0    = blockIdx.x * 128;

  v8f acc[4][2];
#pragma unroll
  for (int i = 0; i < 4; i++)
#pragma unroll
    for (int j = 0; j < 2; j++) acc[i][j] = zero8();

  for (int kk = 0; kk < K; kk += 32) {
    // ---- stage A tile (128x32 fp32 -> bf16), coalesced float4 loads
#pragma unroll
    for (int i = 0; i < 4; i++) {
      int lin = i * 256 + tid;
      int row = lin >> 3;
      int cv  = (lin & 7) * 4;
      const float4 a4 = *(const float4*)(A + (size_t)(m0 + row) * K + kk + cv);
      unsigned p0 = (unsigned)f2bf(a4.x) | ((unsigned)f2bf(a4.y) << 16);
      unsigned p1 = (unsigned)f2bf(a4.z) | ((unsigned)f2bf(a4.w) << 16);
      *(unsigned*)&As[row * 36 + cv]     = p0;
      *(unsigned*)&As[row * 36 + cv + 2] = p1;
    }
    // ---- stage W tile transposed (n-major in global -> [k][n] in LDS)
#pragma unroll
    for (int i = 0; i < 4; i++) {
      int lin = i * 256 + tid;
      int row = lin >> 3;         // n within tile
      int cv  = (lin & 7) * 4;    // k within tile
      const float4 w4 = *(const float4*)(W + (size_t)(n0 + row) * K + kk + cv);
      Wt[(cv + 0) * 132 + row] = f2bf(w4.x);
      Wt[(cv + 1) * 132 + row] = f2bf(w4.y);
      Wt[(cv + 2) * 132 + row] = f2bf(w4.z);
      Wt[(cv + 3) * 132 + row] = f2bf(w4.w);
    }
    __syncthreads();

    // ---- A fragments: lane = M row; K pairs per VGPR (half-wave K+8)
    v16u ar[4], br[2];
#pragma unroll
    for (int mi = 0; mi < 4; mi++) {
      int m = waveM * 64 + mi * 16 + ln;
#pragma unroll
      for (int j = 0; j < 8; j++) {
        int k = kfA(j, half);
        unsigned u = *(const unsigned*)&As[m * 36 + k];
        ar[mi][2 * j]     = (unsigned short)(u & 0xFFFFu);
        ar[mi][2 * j + 1] = (unsigned short)(u >> 16);
      }
    }
    // ---- B fragments: lane = N; K pairs per VGPR (same K order as A)
#pragma unroll
    for (int ni = 0; ni < 2; ni++) {
      int nb = waveN * 32 + ni * 16;
#pragma unroll
      for (int v = 0; v < 8; v++) {
        int kl = kfA(v, half);
        br[ni][2 * v]     = Wt[kl * 132 + nb + ln];
        br[ni][2 * v + 1] = Wt[(kl + 1) * 132 + nb + ln];
      }
    }
#pragma unroll
    for (int mi = 0; mi < 4; mi++) {
      v16bf av = __builtin_bit_cast(v16bf, ar[mi]);
#pragma unroll
      for (int ni = 0; ni < 2; ni++) {
        v16bf bv = __builtin_bit_cast(v16bf, br[ni]);
        acc[mi][ni] = __builtin_amdgcn_wmma_f32_16x16x32_bf16(
            false, av, false, bv, (short)0, acc[mi][ni], false, false);
      }
    }
    __syncthreads();
  }

  // ---- epilogue: C/D layout lane = N, M = r + 8*half
#pragma unroll
  for (int mi = 0; mi < 4; mi++)
#pragma unroll
    for (int ni = 0; ni < 2; ni++)
#pragma unroll
      for (int r = 0; r < 8; r++) {
        int row = m0 + waveM * 64 + mi * 16 + r + 8 * half;
        int col = n0 + waveN * 32 + ni * 16 + ln;
        C[(size_t)row * N + col] = acc[mi][ni][r];
      }
}

// ---------------------------------------------------------------------------
// Kernel 2/4: per-head RMSNorm + RoPE on Q/K, bf16 convert of Q/K/V into
// head-major layouts for the attention kernel. One block = one (b,l,head) row.
// ---------------------------------------------------------------------------
__global__ __launch_bounds__(128) void norm_rope_cvt(
    const float* __restrict__ Q, const float* __restrict__ Kp,
    const float* __restrict__ V, const float* __restrict__ cosb,
    const float* __restrict__ sinb, const float* __restrict__ qw,
    const float* __restrict__ kw, unsigned short* __restrict__ Qr,
    unsigned short* __restrict__ Kr, unsigned short* __restrict__ Vb) {
  __shared__ float red[128];
  const int d  = threadIdx.x;
  const int NH = H_ + 2 * KVH_;
  int bid = blockIdx.x;
  int hv  = bid % NH;
  int l   = (bid / NH) % L_;
  int b   = bid / (NH * L_);
  size_t tok = (size_t)b * L_ + l;

  if (hv < H_ + KVH_) {
    const bool isQ = (hv < H_);
    const int  hh  = isQ ? hv : hv - H_;
    const float* src = isQ ? Q : Kp;
    const float* w   = isQ ? qw : kw;
    const int nh     = isQ ? H_ : KVH_;
    size_t base = tok * (size_t)(nh * HD_) + (size_t)hh * HD_;
    float x = src[base + d];
    int dp = (d < 64) ? d + 64 : d - 64;
    float xp = src[base + dp];

    red[d] = x * x;
    __syncthreads();
#pragma unroll
    for (int s = 64; s > 0; s >>= 1) {
      if (d < s) red[d] += red[d + s];
      __syncthreads();
    }
    float rinv = rsqrtf(red[0] * (1.0f / HD_) + EPS_);
    float xn  = w[d]  * x  * rinv;
    float xpn = w[dp] * xp * rinv;
    float rot = (d < 64) ? -xpn : xpn;
    float cs = cosb[tok * HD_ + d];
    float sn = sinb[tok * HD_ + d];
    float out = xn * cs + rot * sn;
    if (isQ)
      Qr[(((size_t)b * H_ + hh) * L_ + l) * HD_ + d] = f2bf(out);
    else
      Kr[(((size_t)b * KVH_ + hh) * L_ + l) * HD_ + d] = f2bf(out);
  } else {
    int kvh = hv - H_ - KVH_;
    float x = V[tok * (size_t)(KVH_ * HD_) + (size_t)kvh * HD_ + d];
    Vb[(((size_t)b * KVH_ + kvh) * L_ + l) * HD_ + d] = f2bf(x);
  }
}

// ---------------------------------------------------------------------------
// Kernel 3/4: flash attention with online softmax.
// Block: (b, h, 128 q-rows); wave: 16 q-rows; KV streamed in 32-row chunks.
// V tiles staged through the gfx1250 async VMEM->LDS engine (ASYNCcnt);
// K tiles staged manually (needs dim<->kv transpose into LDS).
// Causal mask applied analytically (== reference's triu(NEG) mask).
// ---------------------------------------------------------------------------
__global__ __launch_bounds__(256) void flash_attn(
    const unsigned short* __restrict__ Qr, const unsigned short* __restrict__ Kr,
    const unsigned short* __restrict__ Vb, float* __restrict__ O) {
  __shared__ unsigned short Kt[128 * 36];        // [dim][kv]  (transposed K)
  __shared__ unsigned short Vs[32 * VSTRIDE_];   // [kv][dim]  (row-major V, 16B-aligned rows)
  __shared__ unsigned short Ps[128 * 36];        // per-wave P scratch [16][32]

  const int tid  = threadIdx.x;
  const int lane = tid & 31;
  const int w    = tid >> 5;
  const int half = lane >> 4;
  const int ln   = lane & 15;
  const int qt   = blockIdx.x;
  const int h    = blockIdx.y;
  const int b    = blockIdx.z;
  const int kvh  = h / GROUPS_;
  const int q0   = qt * 128 + w * 16;

  // ---- Q fragments (A layout), held in registers for the whole KV sweep
  v16u aq[4];
  const unsigned* Qp =
      (const unsigned*)(Qr + (((size_t)b * H_ + h) * L_ + q0 + ln) * HD_);
#pragma unroll
  for (int c = 0; c < 4; c++)
#pragma unroll
    for (int j = 0; j < 8; j++) {
      int k = c * 32 + kfA(j, half);
      unsigned u = Qp[k >> 1];
      aq[c][2 * j]     = (unsigned short)(u & 0xFFFFu);
      aq[c][2 * j + 1] = (unsigned short)(u >> 16);
    }

  v8f o[8];
#pragma unroll
  for (int nt = 0; nt < 8; nt++) o[nt] = zero8();
  float mcur[8], lsum[8];
#pragma unroll
  for (int r = 0; r < 8; r++) { mcur[r] = -INFINITY; lsum[r] = 0.0f; }

  const size_t kvbase = ((size_t)b * KVH_ + kvh) * L_ * HD_;
  const int jend = (qt + 1) * 128;

  for (int j = 0; j < jend; j += 32) {
    // ---- V tile: async DMA global->LDS, 16B per lane x2 (B128, ASYNCcnt)
#pragma unroll
    for (int i = 0; i < 2; i++) {
      int lin = i * 256 + tid;
      int row = lin >> 4;              // kv 0..31
      int c   = (lin & 15) * 8;        // dim start (8 bf16 = 16B)
      unsigned long long ga =
          (unsigned long long)(Vb + kvbase + (size_t)(j + row) * HD_ + c);
      unsigned loff =
          (unsigned)(size_t)(&Vs[row * VSTRIDE_ + c]);  // LDS byte offset = addr[31:0]
      asm volatile("global_load_async_to_lds_b128 %0, %1, off"
                   :: "v"(loff), "v"(ga)
                   : "memory");
    }
    // ---- K tile: manual transposed staging [dim][kv]
#pragma unroll
    for (int i = 0; i < 8; i++) {
      int lin = i * 256 + tid;
      int row = lin >> 6;          // kv 0..31
      int dp  = (lin & 63) * 2;    // dim pair
      unsigned ku = *(const unsigned*)(Kr + kvbase + (size_t)(j + row) * HD_ + dp);
      Kt[dp * 36 + row]       = (unsigned short)(ku & 0xFFFFu);
      Kt[(dp + 1) * 36 + row] = (unsigned short)(ku >> 16);
    }
    asm volatile("s_wait_asynccnt 0x0" ::: "memory");
    __syncthreads();

    if (j <= q0 + 15) {  // chunk intersects this wave's causal region
      float smat[2][8];
#pragma unroll
      for (int s2 = 0; s2 < 2; s2++) {
        v8f sc = zero8();
#pragma unroll
        for (int c = 0; c < 4; c++) {
          v16u bk;
#pragma unroll
          for (int v = 0; v < 8; v++) {
            int kl = c * 32 + kfA(v, half);  // head-dim (B-frag lane = N)
            bk[2 * v]     = Kt[kl * 36 + s2 * 16 + ln];
            bk[2 * v + 1] = Kt[(kl + 1) * 36 + s2 * 16 + ln];
          }
          sc = __builtin_amdgcn_wmma_f32_16x16x32_bf16(
              false, __builtin_bit_cast(v16bf, aq[c]), false,
              __builtin_bit_cast(v16bf, bk), (short)0, sc, false, false);
        }
#pragma unroll
        for (int r = 0; r < 8; r++) {
          int col  = j + s2 * 16 + ln;
          int qrow = q0 + r + 8 * half;
          smat[s2][r] = sc[r] * SCALE_ + (col > qrow ? NEG_ : 0.0f);
        }
      }

      // ---- online softmax (row stats reduced across each 16-lane group)
#pragma unroll
      for (int r = 0; r < 8; r++) {
        float m2 = fmaxf(smat[0][r], smat[1][r]);
        m2 = fmaxf(m2, __shfl_xor(m2, 1));
        m2 = fmaxf(m2, __shfl_xor(m2, 2));
        m2 = fmaxf(m2, __shfl_xor(m2, 4));
        m2 = fmaxf(m2, __shfl_xor(m2, 8));
        float mn    = fmaxf(mcur[r], m2);
        float alpha = __expf(mcur[r] - mn);
        float p0 = __expf(smat[0][r] - mn);
        float p1 = __expf(smat[1][r] - mn);
        float rs = p0 + p1;
        rs += __shfl_xor(rs, 1);
        rs += __shfl_xor(rs, 2);
        rs += __shfl_xor(rs, 4);
        rs += __shfl_xor(rs, 8);
        lsum[r] = lsum[r] * alpha + rs;
        mcur[r] = mn;
#pragma unroll
        for (int nt = 0; nt < 8; nt++) o[nt][r] *= alpha;
        // scatter P (C-layout) into wave-private LDS for A-layout reload
        Ps[(w * 16 + r + 8 * half) * 36 + ln]      = f2bf(p0);
        Ps[(w * 16 + r + 8 * half) * 36 + 16 + ln] = f2bf(p1);
      }

      // ---- reload P as A fragment (wave-local LDS, in-order DS, no barrier)
      v16u ap;
#pragma unroll
      for (int jj = 0; jj < 8; jj++) {
        int k = kfA(jj, half);
        unsigned u = *(const unsigned*)&Ps[(w * 16 + ln) * 36 + k];
        ap[2 * jj]     = (unsigned short)(u & 0xFFFFu);
        ap[2 * jj + 1] = (unsigned short)(u >> 16);
      }
      v16bf apv = __builtin_bit_cast(v16bf, ap);

      // ---- O += P * V  (8 HD tiles)
#pragma unroll
      for (int nt = 0; nt < 8; nt++) {
        v16u bvr;
#pragma unroll
        for (int v = 0; v < 8; v++) {
          int kl = kfA(v, half);  // kv index (B-frag lane = N)
          bvr[2 * v]     = Vs[kl * VSTRIDE_ + nt * 16 + ln];
          bvr[2 * v + 1] = Vs[(kl + 1) * VSTRIDE_ + nt * 16 + ln];
        }
        o[nt] = __builtin_amdgcn_wmma_f32_16x16x32_bf16(
            false, apv, false, __builtin_bit_cast(v16bf, bvr), (short)0, o[nt],
            false, false);
      }
    }
    __syncthreads();
  }

  // ---- normalize and write O back to token-major fp32 for the out-projection
#pragma unroll
  for (int r = 0; r < 8; r++) {
    float inv = 1.0f / lsum[r];
    int row = q0 + r + 8 * half;
#pragma unroll
    for (int nt = 0; nt < 8; nt++) {
      O[((size_t)b * L_ + row) * (size_t)(H_ * HD_) + (size_t)h * HD_ +
        nt * 16 + ln] = o[nt][r] * inv;
    }
  }
}

// ---------------------------------------------------------------------------
// Launcher
// Inputs: 0 hidden_states, 1 cos, 2 sin, 3 attention_mask (causal; applied
// analytically in-kernel), 4 wq, 5 wk, 6 wv, 7 wo, 8 q_norm_w, 9 k_norm_w.
// ---------------------------------------------------------------------------
extern "C" void kernel_launch(void* const* d_in, const int* in_sizes, int n_in,
                              void* d_out, int out_size, void* d_ws,
                              size_t ws_size, hipStream_t stream) {
  (void)in_sizes; (void)n_in; (void)out_size; (void)ws_size;
  const float* hidden = (const float*)d_in[0];
  const float* cosb   = (const float*)d_in[1];
  const float* sinb   = (const float*)d_in[2];
  const float* wq     = (const float*)d_in[4];
  const float* wk     = (const float*)d_in[5];
  const float* wv     = (const float*)d_in[6];
  const float* wo     = (const float*)d_in[7];
  const float* qw     = (const float*)d_in[8];
  const float* kw     = (const float*)d_in[9];

  char* ws = (char*)d_ws;
  size_t off = 0;
  float* wsQ = (float*)(ws + off); off += (size_t)4096 * 2048 * 4;
  float* wsK = (float*)(ws + off); off += (size_t)4096 * 512 * 4;
  float* wsV = (float*)(ws + off); off += (size_t)4096 * 512 * 4;
  float* wsO = (float*)(ws + off); off += (size_t)4096 * 2048 * 4;
  unsigned short* Qr = (unsigned short*)(ws + off); off += (size_t)4096 * 2048 * 2;
  unsigned short* Kr = (unsigned short*)(ws + off); off += (size_t)4096 * 512 * 2;
  unsigned short* Vb = (unsigned short*)(ws + off); off += (size_t)4096 * 512 * 2;

  dim3 blk(256);
  // QKV projections
  gemm_xwT<<<dim3(2048 / 128, 4096 / 128), blk, 0, stream>>>(hidden, wq, wsQ,
                                                             4096, 2048, 2048);
  gemm_xwT<<<dim3(512 / 128, 4096 / 128), blk, 0, stream>>>(hidden, wk, wsK,
                                                            4096, 512, 2048);
  gemm_xwT<<<dim3(512 / 128, 4096 / 128), blk, 0, stream>>>(hidden, wv, wsV,
                                                            4096, 512, 2048);
  // RMSNorm + RoPE + bf16 convert
  norm_rope_cvt<<<B_ * L_ * (H_ + 2 * KVH_), 128, 0, stream>>>(
      wsQ, wsK, wsV, cosb, sinb, qw, kw, Qr, Kr, Vb);
  // Flash attention
  flash_attn<<<dim3(L_ / 128, H_, B_), blk, 0, stream>>>(Qr, Kr, Vb, wsO);
  // Output projection
  gemm_xwT<<<dim3(2048 / 128, 4096 / 128), blk, 0, stream>>>(
      wsO, wo, (float*)d_out, 4096, 2048, 2048);
}
